// Synthesis_7919919694265
// MI455X (gfx1250) — compile-verified, matrix operand order checked by reading, not run on hardware
//
#include <hip/hip_runtime.h>

typedef __attribute__((ext_vector_type(16))) __bf16 v16bf;
typedef __attribute__((ext_vector_type(8)))  float  v8f;
typedef __attribute__((ext_vector_type(4)))  unsigned uv4;
typedef __attribute__((ext_vector_type(8)))  unsigned uv8;
typedef unsigned short u16;

__device__ __forceinline__ u16 f2bf(float f) {
    unsigned u = __float_as_uint(f);
    unsigned r = u + 0x7FFFu + ((u >> 16) & 1u);   // round-to-nearest-even
    if ((u & 0x7F800000u) == 0x7F800000u) r = u;   // inf/nan passthrough
    return (u16)(r >> 16);
}

// ---- Tensor Data Mover: 2D tile (rows x 32 bf16) global -> LDS, with LDS row padding ----
// D# per ISA ch8: group0 = {count/lds_addr/global_addr/type}, group1 = dims/strides/pad.
// pad_interval code 3 = 16 DWORDs (64B of data), pad_amount code 3 = 4 DWORDs (16B)
// -> reproduces the 80B LDS row stride of ldsA[..][40].
__device__ __forceinline__ void tdm_load_A(const u16* gsrc, unsigned ldsByteAddr,
                                           int remK, int remRows, int kpadElems) {
    unsigned long long ga = (unsigned long long)(uintptr_t)gsrc;
    uv4 g0;
    g0[0] = 1u;                                               // count=1, user descriptor
    g0[1] = ldsByteAddr;                                      // lds_addr
    g0[2] = (unsigned)(ga & 0xFFFFFFFFull);                   // global_addr[31:0]
    g0[3] = (unsigned)((ga >> 32) & 0x01FFFFFFull) | (2u << 30); // addr[56:32] | type=2
    uv8 g1;
    g1[0] = (1u << 16) | (1u << 20) | (3u << 22) | (3u << 25); // data_size=2B, pad_en, int=16dw, amt=4dw
    g1[1] = ((unsigned)remK & 0xFFFFu) << 16;                 // tensor_dim0[15:0]
    g1[2] = (((unsigned)remK >> 16) & 0xFFFFu)                // tensor_dim0[31:16]
          | (((unsigned)remRows & 0xFFFFu) << 16);            // tensor_dim1[15:0]
    g1[3] = (((unsigned)remRows >> 16) & 0xFFFFu) | (32u << 16); // tile_dim0=32 elems
    g1[4] = 128u;                                             // tile_dim1=128, tile_dim2=0
    g1[5] = (unsigned)kpadElems;                              // tensor_dim0_stride[31:0]
    g1[6] = 0u;                                               // stride hi / dim1_stride lo
    g1[7] = 0u;
    asm volatile("tensor_load_to_lds %0, %1" :: "s"(g0), "s"(g1) : "memory");
}

// ---------------- metric = |fx + fy| at full res ----------------
__global__ void k_metric(const float* __restrict__ flow, float* __restrict__ m,
                         int B, int hwShift) {
    int i = blockIdx.x * blockDim.x + threadIdx.x;
    if (i >= (B << hwShift)) return;
    int b = i >> hwShift, r = i & ((1 << hwShift) - 1);
    float fx = flow[((size_t)(b * 2 + 0) << hwShift) + r];
    float fy = flow[((size_t)(b * 2 + 1) << hwShift) + r];
    m[i] = fabsf(fx + fy);
}

// ---------------- bilinear resize (jax half-pixel), grid.y = B*C ----------------
__global__ void k_resize(const float* __restrict__ in, float* __restrict__ out,
                         int hiShift, int wiShift, int hoShift, int woShift, float mul) {
    int bc = blockIdx.y;
    int i = blockIdx.x * blockDim.x + threadIdx.x;
    int Ho = 1 << hoShift, Wo = 1 << woShift;
    if (i >= (Ho << woShift)) return;
    int Hi = 1 << hiShift, Wi = 1 << wiShift;
    int yo = i >> woShift, xo = i & (Wo - 1);
    float fy = (yo + 0.5f) * (float)Hi / (float)Ho - 0.5f;
    float fx = (xo + 0.5f) * (float)Wi / (float)Wo - 0.5f;
    fy = fminf(fmaxf(fy, 0.0f), (float)(Hi - 1));
    fx = fminf(fmaxf(fx, 0.0f), (float)(Wi - 1));
    int y0 = (int)floorf(fy), x0 = (int)floorf(fx);
    int y1 = min(y0 + 1, Hi - 1), x1 = min(x0 + 1, Wi - 1);
    float ay = fy - (float)y0, ax = fx - (float)x0;
    const float* p = in + ((size_t)bc << (hiShift + wiShift));
    float v00 = p[(y0 << wiShift) + x0], v01 = p[(y0 << wiShift) + x1];
    float v10 = p[(y1 << wiShift) + x0], v11 = p[(y1 << wiShift) + x1];
    float v = (v00 * (1.f - ax) + v01 * ax) * (1.f - ay)
            + (v10 * (1.f - ax) + v11 * ax) * ay;
    out[((size_t)bc << (hoShift + woShift)) + i] = v * mul;
}

// ---------------- zero ----------------
__global__ void k_zero(float* __restrict__ p, size_t n) {
    size_t i = (size_t)blockIdx.x * blockDim.x + threadIdx.x;
    if (i < n) p[i] = 0.0f;
}

// ------- softsplat forward scatter; grid.y = channel (0..c+1) -------
__global__ void k_splat(const float* __restrict__ enc, const float* __restrict__ metric,
                        const float* __restrict__ flow, float* __restrict__ acc,
                        int B, int c, int wShift, int hwShift) {
    int ch = blockIdx.y;
    int chT = c + 2;
    int i = blockIdx.x * blockDim.x + threadIdx.x;
    if (i >= (B << hwShift)) return;
    int b = i >> hwShift, rr = i & ((1 << hwShift) - 1);
    int W = 1 << wShift, H = 1 << (hwShift - wShift);
    int y = rr >> wShift, x = rr & (W - 1);
    float fx = flow[((size_t)(b * 2 + 0) << hwShift) + rr];
    float fy = flow[((size_t)(b * 2 + 1) << hwShift) + rr];
    float m  = metric[((size_t)b << hwShift) + rr];
    float w  = __expf(m);
    float val;
    if (ch < c)       val = enc[((size_t)(b * c + ch) << hwShift) + rr] * w;
    else if (ch == c) val = m * w;
    else              val = w;
    float gx = (float)x + fx, gy = (float)y + fy;
    float x0f = floorf(gx), y0f = floorf(gy);
    int x0 = (int)x0f, y0 = (int)y0f;
    float ax1 = gx - x0f, ay1 = gy - y0f;
    float ax0 = 1.f - ax1, ay0 = 1.f - ay1;
    float* base = acc + ((size_t)(b * chT + ch) << hwShift);
    int   xs[4] = {x0, x0 + 1, x0, x0 + 1};
    int   ys[4] = {y0, y0, y0 + 1, y0 + 1};
    float cw[4] = {ax0 * ay0, ax1 * ay0, ax0 * ay1, ax1 * ay1};
#pragma unroll
    for (int k = 0; k < 4; ++k) {
        int xc = xs[k], yc = ys[k];
        if (xc >= 0 && xc < W && yc >= 0 && yc < H)
            atomicAdd(base + (yc << wShift) + xc, val * cw[k]);
    }
}

// ------- normalize by weight channel -> bf16; grid.y = channel (0..c) -------
__global__ void k_normcvt(const float* __restrict__ acc, u16* __restrict__ X,
                          int B, int c, int hwShift) {
    int ch = blockIdx.y;
    int chT = c + 2;
    int i = blockIdx.x * blockDim.x + threadIdx.x;
    if (i >= (B << hwShift)) return;
    int b = i >> hwShift, rr = i & ((1 << hwShift) - 1);
    float nrm = acc[((size_t)(b * chT + c + 1) << hwShift) + rr];
    if (nrm == 0.0f) nrm = 1.0f;
    float v = acc[((size_t)(b * chT + ch) << hwShift) + rr] / nrm;
    X[((size_t)(b * (c + 1) + ch) << hwShift) + rr] = f2bf(v);
}

// ------- weights f32 OIHW -> bf16 [oc][Kpad]; grid.y = oc -------
__global__ void k_cvtw(const float* __restrict__ w, u16* __restrict__ A,
                       int K, int Kpad) {
    int oc = blockIdx.y;
    int k = blockIdx.x * blockDim.x + threadIdx.x;
    if (k >= Kpad) return;
    A[(size_t)oc * Kpad + k] = (k < K) ? f2bf(w[(size_t)oc * K + k]) : (u16)0;
}

// ---- per-thread im2col gather of 8 consecutive K for one pixel ----
template <int TAPS>
__device__ __forceinline__ void gatherB8(const u16* __restrict__ X, int Cin, int Kreal,
                                         int kBase, int bk0, int bb, int by, int bx,
                                         int wShift, int hwShift, int W, int H,
                                         u16 out[8]) {
#pragma unroll
    for (int e = 0; e < 8; ++e) {
        int kk = kBase + bk0 + e;
        u16 v = 0;
        if (kk < Kreal) {
            int ic  = kk / TAPS;                  // constant divisor
            int tap = kk - ic * TAPS;
            int dy = (TAPS == 9) ? (tap / 3 - 1) : 0;
            int dx = (TAPS == 9) ? (tap % 3 - 1) : 0;
            int yy = by + dy, xx = bx + dx;
            if (yy >= 0 && yy < H && xx >= 0 && xx < W)
                v = X[((size_t)(bb * Cin + ic) << hwShift) + (yy << wShift) + xx];
        }
        out[e] = v;
    }
}

// ---------------- implicit-GEMM conv via v_wmma_f32_16x16x32_bf16 ----------------
// Block = 8 waves. Block tile: 128 oc x 64 px. Wave tile: 16 oc x 64 px (4 wmma/step).
// Double-buffered LDS pipeline: buffer 0 primed with global_load_async_to_lds_b128,
// steady-state A tiles DMA'd by the TDM (tensor_load_to_lds, TENSORcnt).
// MODE 0: f32 + residual ; MODE 1: PReLU + bf16 ; MODE 2: plain f32
template <int TAPS, int MODE>
__global__ void __launch_bounds__(256)
k_conv(const u16* __restrict__ X, const u16* __restrict__ A,
       const float* __restrict__ aprelu, const float* __restrict__ resid,
       u16* __restrict__ obf, float* __restrict__ of32,
       int Cin, int Cout, int wShift, int hwShift, int Kpad, int Kreal) {
    __shared__ __align__(16) u16 ldsA[2][128][40];  // 2 x (128 oc x 32 K), 80B row stride
    __shared__ __align__(16) u16 ldsB[2][64][40];   // 2 x (64 px x 32 K)
    const int tid  = threadIdx.x;
    const int lane = tid & 31;
    const int wave = tid >> 5;
    const int half = lane >> 4;                     // K-half per ISA 16-bit layout
    const int ln   = lane & 15;
    const int W = 1 << wShift, H = 1 << (hwShift - wShift);
    const int hwMask = (1 << hwShift) - 1;
    const int pixBase = blockIdx.x * 64;
    const int ocBase0 = blockIdx.y * 128;
    const int ocBase  = ocBase0 + wave * 16;
    const bool waveValid = (ocBase < Cout);         // Cout multiple of 16

    v8f acc[4] = {v8f{}, v8f{}, v8f{}, v8f{}};
    const int kSteps = Kpad >> 5;

    // this thread's B-staging slot: pixel bp, 8 consecutive K at bk0
    const int bp  = tid >> 2;
    const int bk0 = (tid & 3) * 8;
    const int bpe = pixBase + bp;
    const int bb  = bpe >> hwShift;
    const int brr = bpe & hwMask;
    const int by  = brr >> wShift, bx = brr & (W - 1);

    // ---- prologue: prime buffer 0 (async per-lane copies for A, VALU gather for B) ----
#pragma unroll
    for (int t = 0; t < 2; ++t) {
        int chunk = tid + t * 256;
        int row = chunk >> 2, cno = chunk & 3;
        if (ocBase0 + row < Cout) {
            const u16* src = A + (size_t)(ocBase0 + row) * Kpad + cno * 8;
            unsigned long long ga = (unsigned long long)(uintptr_t)src;
            unsigned dstLds = (unsigned)(uintptr_t)&ldsA[0][row][cno * 8];
            asm volatile("global_load_async_to_lds_b128 %0, %1, off"
                         :: "v"(dstLds), "v"(ga) : "memory");
        }
    }
    {
        union { u16 h[8]; uint4 q; } tv;
        gatherB8<TAPS>(X, Cin, Kreal, 0, bk0, bb, by, bx, wShift, hwShift, W, H, tv.h);
        *(uint4*)&ldsB[0][bp][bk0] = tv.q;
    }
    asm volatile("s_wait_asynccnt 0" ::: "memory");
    __syncthreads();

    for (int ks = 0; ks < kSteps; ++ks) {
        const int cur = ks & 1, nxt = cur ^ 1;
        const bool staging = (ks + 1 < kSteps);
        // ---- stage next buffer: TDM for A tile, VALU gather for B tile ----
        if (staging) {
            const int kN = (ks + 1) << 5;
            if (wave == 0) {
                tdm_load_A(A + (size_t)ocBase0 * Kpad + kN,
                           (unsigned)(uintptr_t)&ldsA[nxt][0][0],
                           Kpad - kN, Cout - ocBase0, Kpad);
            }
            union { u16 h[8]; uint4 q; } tv;
            gatherB8<TAPS>(X, Cin, Kreal, kN, bk0, bb, by, bx, wShift, hwShift, W, H, tv.h);
            *(uint4*)&ldsB[nxt][bp][bk0] = tv.q;
        }
        // ---- consume current buffer: 1 A fragment, 4 B fragments, 4 WMMA ----
        if (waveValid) {
            union U { v16bf v; uint4 q[2]; } ua, ub;
            const u16* arow = &ldsA[cur][wave * 16 + ln][0];
            ua.q[0] = *(const uint4*)(arow + half * 8);
            ua.q[1] = *(const uint4*)(arow + 16 + half * 8);
#pragma unroll
            for (int sub = 0; sub < 4; ++sub) {
                const u16* brow = &ldsB[cur][sub * 16 + ln][0];
                ub.q[0] = *(const uint4*)(brow + half * 8);
                ub.q[1] = *(const uint4*)(brow + 16 + half * 8);
                acc[sub] = __builtin_amdgcn_wmma_f32_16x16x32_bf16(
                               false, ua.v, false, ub.v, (short)0, acc[sub], false, false);
            }
        }
        if (staging) {
            if (wave == 0) asm volatile("s_wait_tensorcnt 0x0" ::: "memory");
            __syncthreads();
        }
    }

    if (!waveValid) return;
#pragma unroll
    for (int sub = 0; sub < 4; ++sub) {
        int pe = pixBase + sub * 16 + ln;           // N = lane & 15
        int b = pe >> hwShift, rr = pe & hwMask;
        size_t obase = ((size_t)b * Cout << hwShift) + rr;
#pragma unroll
        for (int r = 0; r < 8; ++r) {               // M = r + 8*half
            int oc = ocBase + r + 8 * half;
            float v = acc[sub][r];
            size_t oidx = obase + ((size_t)oc << hwShift);
            if (MODE == 1) {
                float al = aprelu[oc];
                v = (v >= 0.0f) ? v : al * v;
                obf[oidx] = f2bf(v);
            } else {
                if (MODE == 0) v += resid[oidx];
                of32[oidx] = v;
            }
        }
    }
}

static inline unsigned ceil_div(unsigned a, unsigned b) { return (a + b - 1) / b; }

extern "C" void kernel_launch(void* const* d_in, const int* in_sizes, int n_in,
                              void* d_out, int out_size, void* d_ws, size_t ws_size,
                              hipStream_t stream) {
    (void)in_sizes; (void)n_in; (void)out_size; (void)ws_size;
    const float* flow0 = (const float*)d_in[1];
    const int B = 4;
    const int dims[4] = {64, 128, 320, 512};

    char* wsb = (char*)d_ws;
    size_t off = 0;
    auto carve = [&](size_t bytes) -> void* {
        void* p = wsb + off;
        off += (bytes + 255) & ~(size_t)255;
        return p;
    };
    float* m_full = (float*)carve((size_t)B * 1024 * 1024 * 4);
    float* m_pp[2] = {(float*)carve((size_t)B * 256 * 256 * 4),
                      (float*)carve((size_t)B * 256 * 256 * 4)};
    float* f_pp[2] = {(float*)carve((size_t)B * 2 * 256 * 256 * 4),
                      (float*)carve((size_t)B * 2 * 256 * 256 * 4)};
    float* accS = (float*)carve((size_t)B * 66 * 256 * 256 * 4); // splat acc / shortcut buf
    u16*   Xbf  = (u16*)carve((size_t)B * 65 * 256 * 256 * 2);
    u16*   T1   = (u16*)carve((size_t)B * 64 * 256 * 256 * 2);
    u16*   A1   = (u16*)carve((size_t)512 * 4640 * 2);
    u16*   A2   = (u16*)carve((size_t)512 * 4608 * 2);
    u16*   As   = (u16*)carve((size_t)512 * 544 * 2);

    k_metric<<<ceil_div(B << 20, 256), 256, 0, stream>>>(flow0, m_full, B, 20);

    const float* mPrev = m_full; int mhS = 10, mwS = 10;
    const float* fPrev = flow0;  int fhS = 10, fwS = 10;
    float* outp = (float*)d_out;
    size_t outOff = 0;

    for (int i = 0; i < 4; ++i) {
        int c = dims[i];
        int wS = 8 - i, hS = 8 - i, hwS = wS + hS;
        int Npix = B << hwS;
        const float* enc = (const float*)d_in[2 + 5 * i];
        const float* w1  = (const float*)d_in[3 + 5 * i];
        const float* ap  = (const float*)d_in[4 + 5 * i];
        const float* w2  = (const float*)d_in[5 + 5 * i];
        const float* wsc = (const float*)d_in[6 + 5 * i];
        float* mCur = m_pp[i & 1];
        float* fCur = f_pp[i & 1];
        float scale = (float)(1 << wS) / (float)(1 << fwS);

        {
            dim3 g(ceil_div(1u << hwS, 256), B * 1);
            k_resize<<<g, 256, 0, stream>>>(mPrev, mCur, mhS, mwS, hS, wS, 1.0f);
        }
        {
            dim3 g(ceil_div(1u << hwS, 256), B * 2);
            k_resize<<<g, 256, 0, stream>>>(fPrev, fCur, fhS, fwS, hS, wS, scale);
        }

        int chT = c + 2;
        size_t accN = (size_t)chT * Npix;
        k_zero<<<ceil_div((unsigned)accN, 256), 256, 0, stream>>>(accS, accN);
        {
            dim3 g(ceil_div((unsigned)Npix, 256), chT);
            k_splat<<<g, 256, 0, stream>>>(enc, mCur, fCur, accS, B, c, wS, hwS);
        }
        {
            dim3 g(ceil_div((unsigned)Npix, 256), c + 1);
            k_normcvt<<<g, 256, 0, stream>>>(accS, Xbf, B, c, hwS);
        }

        int K1 = (c + 1) * 9, K1p = (K1 + 31) & ~31;
        int K2 = c * 9,       K2p = (K2 + 31) & ~31;
        int Ks = c + 1,       Ksp = (Ks + 31) & ~31;
        { dim3 g(ceil_div((unsigned)K1p, 256), c); k_cvtw<<<g, 256, 0, stream>>>(w1, A1, K1, K1p); }
        { dim3 g(ceil_div((unsigned)K2p, 256), c); k_cvtw<<<g, 256, 0, stream>>>(w2, A2, K2, K2p); }
        { dim3 g(ceil_div((unsigned)Ksp, 256), c); k_cvtw<<<g, 256, 0, stream>>>(wsc, As, Ks, Ksp); }

        dim3 blk(256);
        dim3 grd(ceil_div((unsigned)Npix, 64), ceil_div((unsigned)c, 128));
        // shortcut 1x1: X(c+1) -> accS (f32)
        k_conv<1, 2><<<grd, blk, 0, stream>>>(Xbf, As, nullptr, nullptr, nullptr, accS,
                                              c + 1, c, wS, hwS, Ksp, Ks);
        // conv1 3x3 + PReLU: X(c+1) -> T1 (bf16)
        k_conv<9, 1><<<grd, blk, 0, stream>>>(Xbf, A1, ap, nullptr, T1, nullptr,
                                              c + 1, c, wS, hwS, K1p, K1);
        // conv2 3x3 + residual: T1(c) -> out (f32)
        k_conv<9, 0><<<grd, blk, 0, stream>>>(T1, A2, nullptr, accS, nullptr, outp + outOff,
                                              c, c, wS, hwS, K2p, K2);
        outOff += (size_t)c * Npix;

        mPrev = mCur; mhS = hS; mwS = wS;
        fPrev = fCur; fhS = hS; fwS = wS;
    }
}
